// MultiheadStaticAttention_11570641895992
// MI455X (gfx1250) — matrix operand order repaired and checked by reading, not compiled, not run on hardware
//
#include <hip/hip_runtime.h>

typedef __bf16 bf16_t;
typedef __attribute__((ext_vector_type(16))) __bf16 bf16x16;
typedef __attribute__((ext_vector_type(8)))  __bf16 bf16x8;
typedef __attribute__((ext_vector_type(4)))  __bf16 bf16x4;
typedef __attribute__((ext_vector_type(8)))  float  f32x8;
typedef __attribute__((ext_vector_type(4)))  float  f32x4;
typedef __attribute__((ext_vector_type(4)))  unsigned int u32x4;
typedef __attribute__((ext_vector_type(8)))  int    i32x8;
typedef __attribute__((ext_vector_type(4)))  int    i32x4;

#define WMMA_BF16(a, b, c) \
  __builtin_amdgcn_wmma_f32_16x16x32_bf16(false, (a), false, (b), (short)0, (c), false, false)

#if defined(__gfx1250__) && __has_builtin(__builtin_amdgcn_tensor_load_to_lds) && \
    __has_builtin(__builtin_amdgcn_s_wait_tensorcnt)
#define USE_TDM 1
#else
#define USE_TDM 0
#endif

constexpr int kB   = 8;
constexpr int kL   = 2048;
constexpr int kD   = 768;
constexpr int kH   = 16;
constexpr int kDh  = 48;
constexpr int kE   = 300;
constexpr int kEp  = 320;   // E padded to multiple of 32
constexpr int kDhp = 64;    // dh padded to multiple of 32
constexpr int kM   = kB * kL;

// ---------------------------------------------------------------------------
// Fragment loaders (CDNA5 16x16x32 bf16, wave32)
// ---------------------------------------------------------------------------
__device__ __forceinline__ bf16x16 load_frag_a(const bf16_t* __restrict__ src, int ld) {
  int lane = threadIdx.x & 31;
  int m  = lane & 15;
  int kb = (lane >> 4) << 3;
  bf16x16 f;
#pragma unroll
  for (int i = 0; i < 16; ++i) {
    int k = kb + (i & 7) + ((i & 8) << 1);
    f[i] = src[m * ld + k];
  }
  return f;
}

// B fragment from [n][k]-major storage: table k order (matches A fragments)
__device__ __forceinline__ bf16x16 load_frag_bT(const bf16_t* __restrict__ src, int ld) {
  int lane = threadIdx.x & 31;
  int n  = lane & 15;
  int kb = (lane >> 4) << 3;
  bf16x16 f;
#pragma unroll
  for (int i = 0; i < 16; ++i) f[i] = src[n * ld + kb + (i & 7) + ((i & 8) << 1)];
  return f;
}

// A fragment from global bf16 Q rows, zero-padded beyond kmax (dh=48)
__device__ __forceinline__ bf16x16 load_frag_a_gq(const bf16_t* __restrict__ rowbase,
                                                  int ld, int kd, int kmax) {
  int lane = threadIdx.x & 31;
  int m  = lane & 15;
  int kb = (lane >> 4) << 3;
  bf16x16 f;
#pragma unroll
  for (int i = 0; i < 16; ++i) {
    int k = kd + kb + (i & 7) + ((i & 8) << 1);
    f[i] = (k < kmax) ? rowbase[m * ld + k] : (bf16_t)0.0f;
  }
  return f;
}

// ---------------------------------------------------------------------------
// Kernel 0: pad + convert memory (E,H,dh) f32 -> memA[h][320][64] bf16
// ---------------------------------------------------------------------------
__global__ void prep_mem_kernel(const float* __restrict__ mem, bf16_t* __restrict__ memA) {
  int i = blockIdx.x * 256 + threadIdx.x;
  if (i >= kH * kEp * kDhp) return;
  int d = i & (kDhp - 1);
  int e = (i >> 6) % kEp;
  int h = i / (kEp * kDhp);
  float v = 0.0f;
  if (e < kE && d < kDh) v = mem[(e * kH + h) * kDh + d];
  memA[i] = (bf16_t)v;
}

// ---------------------------------------------------------------------------
// Kernel 1: projection GEMM  Out = bf16(A(16384x768) * W(768x768) + bias)
// Double-buffered LDS; batched global->reg loads; one barrier per K-step.
// ---------------------------------------------------------------------------
__global__ __launch_bounds__(256) void proj_gemm_kernel(
    const float* __restrict__ A, const float* __restrict__ W,
    const float* __restrict__ bias, bf16_t* __restrict__ Out) {
  constexpr int TM = 128, TN = 64, TK = 32;
  constexpr int LDA = TK + 8;   // 40 elements (80B rows, 16B aligned)
  constexpr int LDB = TK + 8;
  constexpr int NK = kD / TK;   // 24
  __shared__ bf16_t As[2][TM][LDA];
  __shared__ bf16_t Bs[2][TN][LDB];

  const int m0 = blockIdx.y * TM;
  const int n0 = blockIdx.x * TN;
  const int t  = threadIdx.x;
  const int w  = t >> 5;
  const int wm = (w >> 1) * 32;
  const int wn = (w & 1) * 32;

  f32x8 acc[2][2];
#pragma unroll
  for (int i = 0; i < 2; ++i)
#pragma unroll
    for (int j = 0; j < 2; ++j) acc[i][j] = (f32x8){0.f,0.f,0.f,0.f,0.f,0.f,0.f,0.f};

  f32x4 ra[4], rb[2];

  // ---- load phase: batched, no intervening waits ----
  auto tile_load = [&](int k0) {
#pragma unroll
    for (int it = 0; it < 4; ++it) {
      int i = t + it * 256;
      ra[it] = *(const f32x4*)&A[(size_t)(m0 + (i >> 3)) * kD + k0 + (i & 7) * 4];
    }
#pragma unroll
    for (int it = 0; it < 2; ++it) {
      int i = t + it * 256;
      rb[it] = *(const f32x4*)&W[(size_t)(k0 + (i >> 4)) * kD + n0 + (i & 15) * 4];
    }
  };
  // ---- store phase: convert + LDS ----
  auto tile_store = [&](int buf) {
#pragma unroll
    for (int it = 0; it < 4; ++it) {
      int i = t + it * 256;
      int r = i >> 3, c4 = (i & 7) * 4;
      bf16x4 o;
      o[0] = (bf16_t)ra[it][0]; o[1] = (bf16_t)ra[it][1];
      o[2] = (bf16_t)ra[it][2]; o[3] = (bf16_t)ra[it][3];
      *(bf16x4*)&As[buf][r][c4] = o;
    }
#pragma unroll
    for (int it = 0; it < 2; ++it) {
      int i = t + it * 256;
      int r = i >> 4, c4 = (i & 15) * 4;
      Bs[buf][c4 + 0][r] = (bf16_t)rb[it][0];
      Bs[buf][c4 + 1][r] = (bf16_t)rb[it][1];
      Bs[buf][c4 + 2][r] = (bf16_t)rb[it][2];
      Bs[buf][c4 + 3][r] = (bf16_t)rb[it][3];
    }
  };

  tile_load(0);
  tile_store(0);
  __syncthreads();

  for (int ki = 0; ki < NK; ++ki) {
    const int cur = ki & 1, nxt = cur ^ 1;
    if (ki + 1 < NK) tile_load((ki + 1) * TK);      // in flight during WMMA
    if (ki + 2 < NK)                                 // warm L2 two tiles ahead
      __builtin_prefetch(&A[(size_t)(m0 + (t >> 1)) * kD + (ki + 2) * TK], 0, 3);

    bf16x16 a0 = load_frag_a(&As[cur][wm][0],      LDA);
    bf16x16 a1 = load_frag_a(&As[cur][wm + 16][0], LDA);
    bf16x16 b0 = load_frag_bT(&Bs[cur][wn][0],      LDB);
    bf16x16 b1 = load_frag_bT(&Bs[cur][wn + 16][0], LDB);
    acc[0][0] = WMMA_BF16(a0, b0, acc[0][0]);
    acc[0][1] = WMMA_BF16(a0, b1, acc[0][1]);
    acc[1][0] = WMMA_BF16(a1, b0, acc[1][0]);
    acc[1][1] = WMMA_BF16(a1, b1, acc[1][1]);

    if (ki + 1 < NK) tile_store(nxt);
    __syncthreads();
  }

  const int lane = threadIdx.x & 31;
  const int n  = lane & 15;
  const int mo = (lane >> 4) << 3;
#pragma unroll
  for (int im = 0; im < 2; ++im)
#pragma unroll
    for (int in = 0; in < 2; ++in) {
      int gn = n0 + wn + in * 16 + n;
      float bv = bias[gn];
#pragma unroll
      for (int r = 0; r < 8; ++r) {
        int gm = m0 + wm + im * 16 + mo + r;
        Out[(size_t)gm * kD + gn] = (bf16_t)(acc[im][in][r] + bv);
      }
    }
}

// ---------------------------------------------------------------------------
// Kernel 2 (pass A): out1[b,h] = sum_L relu(M_h * Q^T) * V   (E x dh, padded)
// Ms via TDM (padded rows); double-buffered Q/V tiles over the L loop.
// out1 written TRANSPOSED: out1T[b][h][d][e].
// ---------------------------------------------------------------------------
__global__ __launch_bounds__(320) void pass_a_kernel(
    const bf16_t* __restrict__ Qbf, const bf16_t* __restrict__ Vbf,
    const bf16_t* __restrict__ memA, bf16_t* __restrict__ out1T) {
  constexpr int LDM = kDhp + 8;   // 72 (144B rows)
  constexpr int LDQ = kDhp + 8;   // Qt[l][d] = natural Q rows ([n][k] for B-frags)
  constexpr int LDV = 40;         // Vt[d][l] ([n][k] for B-frags)
  constexpr int NLT = kL / 32;    // 64 L-tiles
  __shared__ bf16_t Ms[kEp][LDM];
  __shared__ bf16_t Qt[2][32][LDQ];
  __shared__ bf16_t Vt[2][kDhp][LDV];
  __shared__ bf16_t Sb[10][16][40];

  const int b = blockIdx.x;
  const int h = blockIdx.y;
  const int t = threadIdx.x;
  const int w = t >> 5;
  const int lane = t & 31;
  const int n_ = lane & 15;
  const int mo = (lane >> 4) << 3;

#if USE_TDM
  if (w == 0) {
    unsigned lds_base = (unsigned)(size_t)(void*)&Ms[0][0];
    unsigned long long ga =
        (unsigned long long)(size_t)(memA + (size_t)h * kEp * kDhp);
    u32x4 g0;
    g0[0] = 1u;                                   // count=1 (valid user D#)
    g0[1] = lds_base;                             // lds_addr
    g0[2] = (unsigned)(ga & 0xffffffffu);         // global_addr[31:0]
    g0[3] = (unsigned)((ga >> 32) & 0x01ffffffu)  // global_addr[56:32]
            | 0x80000000u;                        // type = 2 ("image")
    i32x8 g1;
    g1[0] = (1 << 16)      // data_size = 2 bytes
          | (1 << 20)      // pad_enable
          | (4 << 22)      // pad_interval: 32 DWORDs (=128B row)
          | (3 << 25);     // pad_amount: 4 DWORDs (=16B)
    g1[1] = (kDhp & 0xffff) << 16;                            // tensor_dim0 lo16
    g1[2] = ((kDhp >> 16) & 0xffff) | ((kEp & 0xffff) << 16); // dim0 hi | dim1 lo
    g1[3] = ((kEp >> 16) & 0xffff) | (kDhp << 16);            // dim1 hi | tile_dim0
    g1[4] = (kEp & 0xffff);                                   // tile_dim1 | tile_dim2=0
    g1[5] = kDhp;                                             // dim0_stride lo32
    g1[6] = 0;
    g1[7] = 0;
    i32x4 z4 = (i32x4){0, 0, 0, 0};
#if __clang_major__ >= 23
    i32x8 z8 = (i32x8){0, 0, 0, 0, 0, 0, 0, 0};
    __builtin_amdgcn_tensor_load_to_lds(g0, g1, z4, z4, z8, 0);
#else
    __builtin_amdgcn_tensor_load_to_lds(g0, g1, z4, z4, 0);
#endif
  }
  __builtin_amdgcn_s_wait_tensorcnt(0);
  __syncthreads();
#else
#pragma unroll
  for (int it = 0; it < 8; ++it) {
    int i = t + it * 320;
    if (i < kEp * (kDhp / 8)) {
      int e = i >> 3, d8 = (i & 7) * 8;
      *(bf16x8*)&Ms[e][d8] =
          *(const bf16x8*)&memA[((size_t)h * kEp + e) * kDhp + d8];
    }
  }
  __syncthreads();
#endif

  // hoisted A-fragments of M_h
  bf16x16 aM[2][2];
#pragma unroll
  for (int et = 0; et < 2; ++et)
#pragma unroll
    for (int kd = 0; kd < 2; ++kd)
      aM[et][kd] = load_frag_a(&Ms[w * 32 + et * 16][kd * 32], LDM);

  f32x8 o1[2][4];
#pragma unroll
  for (int et = 0; et < 2; ++et)
#pragma unroll
    for (int j = 0; j < 4; ++j) o1[et][j] = (f32x8){0.f,0.f,0.f,0.f,0.f,0.f,0.f,0.f};

  // double-buffered Q/V staging (256 active staging threads)
  const int sl  = t >> 3;
  const int sd8 = (t & 7) * 8;
  bf16x8 rq{}, rv{};
  auto qv_load = [&](int lt) {
    rq = bf16x8{}; rv = bf16x8{};
    if (t < 256 && sd8 < kDh) {
      size_t row = (size_t)b * kL + lt + sl;
      rq = *(const bf16x8*)&Qbf[row * kD + h * kDh + sd8];
      rv = *(const bf16x8*)&Vbf[row * kD + h * kDh + sd8];
    }
  };
  auto qv_store = [&](int buf) {
    if (t < 256) {
      *(bf16x8*)&Qt[buf][sl][sd8] = rq;
#pragma unroll
      for (int j2 = 0; j2 < 8; ++j2) Vt[buf][sd8 + j2][sl] = rv[j2];
    }
  };

  qv_load(0);
  qv_store(0);
  __syncthreads();

  for (int li = 0; li < NLT; ++li) {
    const int cur = li & 1, nxt = cur ^ 1;
    if (li + 1 < NLT) qv_load((li + 1) * 32);

    bf16x16 bq0 = load_frag_bT(&Qt[cur][0][0],   LDQ);
    bf16x16 bq1 = load_frag_bT(&Qt[cur][16][0],  LDQ);
    bf16x16 bq2 = load_frag_bT(&Qt[cur][0][32],  LDQ);
    bf16x16 bq3 = load_frag_bT(&Qt[cur][16][32], LDQ);
    bf16x16 bv[4];
#pragma unroll
    for (int j = 0; j < 4; ++j) bv[j] = load_frag_bT(&Vt[cur][j * 16][0], LDV);

#pragma unroll
    for (int et = 0; et < 2; ++et) {
      f32x8 s0 = (f32x8){0.f,0.f,0.f,0.f,0.f,0.f,0.f,0.f};
      f32x8 s1 = s0;
      s0 = WMMA_BF16(aM[et][0], bq0, s0);
      s1 = WMMA_BF16(aM[et][0], bq1, s1);
      s0 = WMMA_BF16(aM[et][1], bq2, s0);
      s1 = WMMA_BF16(aM[et][1], bq3, s1);

#pragma unroll
      for (int r = 0; r < 8; ++r) {
        Sb[w][mo + r][n_]      = (bf16_t)fmaxf(s0[r], 0.0f);
        Sb[w][mo + r][16 + n_] = (bf16_t)fmaxf(s1[r], 0.0f);
      }
      asm volatile("s_wait_dscnt 0" ::: "memory");
      bf16x16 aS = load_frag_a(&Sb[w][0][0], 40);

#pragma unroll
      for (int j = 0; j < 4; ++j) o1[et][j] = WMMA_BF16(aS, bv[j], o1[et][j]);
    }

    if (li + 1 < NLT) qv_store(nxt);
    __syncthreads();
  }

  bf16_t* o = out1T + ((size_t)(b * kH + h)) * kDhp * kEp;
#pragma unroll
  for (int et = 0; et < 2; ++et)
#pragma unroll
    for (int j = 0; j < 4; ++j)
#pragma unroll
      for (int r = 0; r < 8; ++r) {
        int e = w * 32 + et * 16 + mo + r;
        int d = j * 16 + n_;
        o[(size_t)d * kEp + e] = (bf16_t)o1[et][j][r];
      }
}

// ---------------------------------------------------------------------------
// Kernel 3 (pass B + LayerNorm): out2 = relu(Q*M_h^T) * out1, then LN over D.
// ---------------------------------------------------------------------------
__global__ __launch_bounds__(256) void pass_b_ln_kernel(
    const bf16_t* __restrict__ Qbf, const bf16_t* __restrict__ memA,
    const bf16_t* __restrict__ out1T, const float* __restrict__ gamma,
    const float* __restrict__ beta, float* __restrict__ out) {
  __shared__ float  X[16][kD + 16];
  __shared__ bf16_t Tb[8][16][40];

  const int lrow0 = blockIdx.x * 16;
  const int b     = blockIdx.y;
  const int w     = threadIdx.x >> 5;
  const int lane  = threadIdx.x & 31;
  const int n_    = lane & 15;
  const int mo    = (lane >> 4) << 3;

#pragma unroll
  for (int hh = 0; hh < 2; ++hh) {
    const int h = w + hh * 8;
    const bf16_t* qrow = Qbf + ((size_t)b * kL + lrow0) * kD + h * kDh;
    const bf16_t* mh   = memA + (size_t)h * kEp * kDhp;               // [e][d]
    const bf16_t* o1h  = out1T + ((size_t)(b * kH + h)) * kDhp * kEp; // [d][e]

    f32x8 oacc[4];
#pragma unroll
    for (int j = 0; j < 4; ++j) oacc[j] = (f32x8){0.f,0.f,0.f,0.f,0.f,0.f,0.f,0.f};

    bf16x16 aq0 = load_frag_a_gq(qrow, kD, 0,  kDh);
    bf16x16 aq1 = load_frag_a_gq(qrow, kD, 32, kDh);

    for (int ec = 0; ec < kEp; ec += 32) {
      f32x8 t0 = (f32x8){0.f,0.f,0.f,0.f,0.f,0.f,0.f,0.f};
      f32x8 t1 = t0;
      t0 = WMMA_BF16(aq0, load_frag_bT(mh + (size_t)ec * kDhp,        kDhp), t0);
      t1 = WMMA_BF16(aq0, load_frag_bT(mh + (size_t)(ec + 16) * kDhp, kDhp), t1);
      t0 = WMMA_BF16(aq1, load_frag_bT(mh + (size_t)ec * kDhp + 32,   kDhp), t0);
      t1 = WMMA_BF16(aq1, load_frag_bT(mh + (size_t)(ec + 16) * kDhp + 32, kDhp), t1);

#pragma unroll
      for (int r = 0; r < 8; ++r) {
        Tb[w][mo + r][n_]      = (bf16_t)fmaxf(t0[r], 0.0f);
        Tb[w][mo + r][16 + n_] = (bf16_t)fmaxf(t1[r], 0.0f);
      }
      asm volatile("s_wait_dscnt 0" ::: "memory");
      bf16x16 aT = load_frag_a(&Tb[w][0][0], 40);

#pragma unroll
      for (int j = 0; j < 4; ++j)
        oacc[j] = WMMA_BF16(aT, load_frag_bT(o1h + (size_t)(j * 16) * kEp + ec, kEp),
                            oacc[j]);
    }

#pragma unroll
    for (int j = 0; j < 4; ++j) {
      int dl = j * 16 + n_;
      if (dl < kDh) {
#pragma unroll
        for (int r = 0; r < 8; ++r) X[mo + r][h * kDh + dl] = oacc[j][r];
      }
    }
  }
  __syncthreads();

#pragma unroll
  for (int rr = w * 2; rr < w * 2 + 2; ++rr) {
    float s = 0.0f, s2 = 0.0f;
    for (int c = lane; c < kD; c += 32) {
      float x = X[rr][c];
      s += x; s2 += x * x;
    }
#pragma unroll
    for (int off = 16; off > 0; off >>= 1) {
      s  += __shfl_xor(s, off, 32);
      s2 += __shfl_xor(s2, off, 32);
    }
    float mean = s * (1.0f / kD);
    float var  = s2 * (1.0f / kD) - mean * mean;
    float inv  = rsqrtf(var + 1e-3f);
    float* orow = out + ((size_t)b * kL + lrow0 + rr) * kD;
    for (int c = lane; c < kD; c += 32) {
      orow[c] = (X[rr][c] - mean) * inv * gamma[c] + beta[c];
    }
  }
}

// ---------------------------------------------------------------------------
// Launch
// ---------------------------------------------------------------------------
extern "C" void kernel_launch(void* const* d_in, const int* in_sizes, int n_in,
                              void* d_out, int out_size, void* d_ws, size_t ws_size,
                              hipStream_t stream) {
  (void)in_sizes; (void)n_in; (void)out_size; (void)ws_size;

  const float* q     = (const float*)d_in[0];
  const float* k     = (const float*)d_in[1];
  const float* Wq    = (const float*)d_in[3];
  const float* bq    = (const float*)d_in[4];
  const float* Wv    = (const float*)d_in[5];
  const float* bvp   = (const float*)d_in[6];
  const float* mem   = (const float*)d_in[7];
  const float* gamma = (const float*)d_in[8];
  const float* beta  = (const float*)d_in[9];
  float* out = (float*)d_out;

  const size_t qbf_off  = 0;
  const size_t vbf_off  = qbf_off + (size_t)kM * kD * sizeof(bf16_t);
  const size_t mem_off  = vbf_off + (size_t)kM * kD * sizeof(bf16_t);
  const size_t out1_off = mem_off + (size_t)kH * kEp * kDhp * sizeof(bf16_t);
  char* ws = (char*)d_ws;
  bf16_t* Qbf   = (bf16_t*)(ws + qbf_off);
  bf16_t* Vbf   = (bf16_t*)(ws + vbf_off);
  bf16_t* memA  = (bf16_t*)(ws + mem_off);
  bf16_t* out1T = (bf16_t*)(ws + out1_off);

  {
    int total = kH * kEp * kDhp;
    prep_mem_kernel<<<(total + 255) / 256, 256, 0, stream>>>(mem, memA);
  }
  {
    dim3 grid(kD / 64, kM / 128);
    proj_gemm_kernel<<<grid, 256, 0, stream>>>(q, Wq, bq, Qbf);
    proj_gemm_kernel<<<grid, 256, 0, stream>>>(k, Wv, bvp, Vbf);
  }
  {
    dim3 grid(kB, kH);
    pass_a_kernel<<<grid, 320, 0, stream>>>(Qbf, Vbf, memA, out1T);
  }
  {
    dim3 grid(kL / 16, kB);
    pass_b_ln_kernel<<<grid, 256, 0, stream>>>(Qbf, memA, out1T, gamma, beta, out);
  }
}
